// SGNS_58772332478762
// MI455X (gfx1250) — matrix operand order repaired and checked by reading, not compiled, not run on hardware
//
#include <hip/hip_runtime.h>

typedef __attribute__((ext_vector_type(16))) _Float16 v16h;
typedef __attribute__((ext_vector_type(8)))  float    v8f;

#define DIM      300
#define BATCH    4096
#define CTX      10
#define NNEG     200                    // CTX * N_NEGS
#define NSLOT    210                    // CTX + NNEG
#define NCHUNK   10                     // ceil(DIM / 32)
#define WAVES    8                      // 256-thread block = 8 wave32
#define YSPLIT   4                      // slot-dimension grid split
#define SLOTSTRIDE (WAVES * YSPLIT)     // 32 slot workers per batch tile
#define NPART    ((BATCH / 16) * YSPLIT * WAVES)   // 8192 partials

__device__ __forceinline__ float log_sigmoid_f(float x) {
    // logσ(x) = min(x,0) - log1p(exp(-|x|)); 1+e in (1,2] so raw HW trans ops suffice
    float ax = __builtin_fabsf(x);
    float e  = __builtin_amdgcn_exp2f(ax * -1.4426950408889634f);     // v_exp_f32
    float l  = __builtin_amdgcn_logf(1.0f + e) * 0.6931471805599453f; // v_log_f32
    return fminf(x, 0.0f) - l;
}

template <int BASE>
__device__ __forceinline__ void pack8(v16h& v, float4 x, float4 y) {
    v[BASE + 0] = (_Float16)x.x; v[BASE + 1] = (_Float16)x.y;
    v[BASE + 2] = (_Float16)x.z; v[BASE + 3] = (_Float16)x.w;
    v[BASE + 4] = (_Float16)y.x; v[BASE + 5] = (_Float16)y.y;
    v[BASE + 6] = (_Float16)y.z; v[BASE + 7] = (_Float16)y.w;
}

__device__ __forceinline__ void zero16(v16h& v) {
    #pragma unroll
    for (int e = 0; e < 16; ++e) v[e] = (_Float16)0.0f;
}

__global__ __launch_bounds__(256)
void sgns_score_kernel(const int* __restrict__ iword,
                       const int* __restrict__ owords,
                       const int* __restrict__ nwords,
                       const float* __restrict__ emb_i,
                       const float* __restrict__ emb_o,
                       float* __restrict__ partials)
{
    const int lane  = threadIdx.x & 31;
    const int wave  = threadIdx.x >> 5;
    const int bbase = blockIdx.x << 4;      // 16 batch rows per tile
    const int col   = lane & 15;            // B-matrix column = batch sub-index
    const bool hi   = lane >= 16;
    const int  b    = bbase + col;

    // ---- A fragments: ivectors for this 16-row tile (reused for all slots) ----
    // 16-bit A 16x32 layout: lanes 0-15 K={0..7,16..23}; lanes 16-31 K={8..15,24..31}
    const float* arow = emb_i + (size_t)iword[b] * DIM;
    v16h a[NCHUNK];
    #pragma unroll
    for (int kc = 0; kc < NCHUNK - 1; ++kc) {
        const int k0 = kc * 32 + (hi ? 8 : 0);
        const int k1 = kc * 32 + (hi ? 24 : 16);
        float4 f0 = *(const float4*)(arow + k0);
        float4 f1 = *(const float4*)(arow + k0 + 4);
        float4 f2 = *(const float4*)(arow + k1);
        float4 f3 = *(const float4*)(arow + k1 + 4);
        pack8<0>(a[kc], f0, f1);
        pack8<8>(a[kc], f2, f3);
    }
    // tail chunk kc=9: dims 288..299 valid, rest zero-padded
    zero16(a[NCHUNK - 1]);
    if (!hi) {                               // K=288..295 -> elements 0..7
        float4 f0 = *(const float4*)(arow + 288);
        float4 f1 = *(const float4*)(arow + 292);
        pack8<0>(a[NCHUNK - 1], f0, f1);
    } else {                                 // K=296..299 -> elements 0..3
        float4 f0 = *(const float4*)(arow + 296);
        a[NCHUNK - 1][0] = (_Float16)f0.x; a[NCHUNK - 1][1] = (_Float16)f0.y;
        a[NCHUNK - 1][2] = (_Float16)f0.z; a[NCHUNK - 1][3] = (_Float16)f0.w;
    }

    float wsum = 0.0f;
    const int worker = wave + WAVES * blockIdx.y;

    for (int s = worker; s < NSLOT; s += SLOTSTRIDE) {
        int   widx;
        float sgn;
        if (s < CTX) { widx = owords[b * CTX + s];          sgn =  1.0f; }
        else         { widx = nwords[b * NNEG + (s - CTX)]; sgn = -1.0f; }
        const float* orow = emb_o + (size_t)widx * DIM;

        // Issue next slot's index load NOW so it's in flight alongside the row
        // loads (loads complete in order -> value is free once rows arrive).
        const int s2 = s + SLOTSTRIDE;
        int w2 = 0;
        const bool have_next = (s2 < NSLOT);
        if (have_next)
            w2 = (s2 < CTX) ? owords[b * CTX + s2]
                            : nwords[b * NNEG + (s2 - CTX)];

        v8f acc = {0.f, 0.f, 0.f, 0.f, 0.f, 0.f, 0.f, 0.f};
        #pragma unroll
        for (int kc = 0; kc < NCHUNK; ++kc) {
            // 16-bit B 32x16 layout: lanes 0-15 K={0..7,16..23}; lanes 16-31 K={8..15,24..31}
            const int kb = kc * 32 + (hi ? 8 : 0);
            v16h bf;
            if (kc < NCHUNK - 1) {
                float4 g0 = *(const float4*)(orow + kb);
                float4 g1 = *(const float4*)(orow + kb + 4);
                float4 g2 = *(const float4*)(orow + kb + 16);
                float4 g3 = *(const float4*)(orow + kb + 20);
                pack8<0>(bf, g0, g1);
                pack8<8>(bf, g2, g3);
            } else {
                zero16(bf);
                if (!hi) {                   // K=288..295 -> elements 0..7
                    float4 g0 = *(const float4*)(orow + 288);
                    float4 g1 = *(const float4*)(orow + 292);
                    pack8<0>(bf, g0, g1);
                } else {                     // K=296..299 -> elements 0..3
                    float4 g2 = *(const float4*)(orow + 296);
                    bf[0] = (_Float16)g2.x; bf[1] = (_Float16)g2.y;
                    bf[2] = (_Float16)g2.z; bf[3] = (_Float16)g2.w;
                }
            }
            acc = __builtin_amdgcn_wmma_f32_16x16x32_f16(
                false, a[kc], false, bf, (short)0, acc, false, false);
        }

        // Prefetch next slot's gathered row (10 x 128B cachelines cover 1200B)
        // after the current row loads have been issued; near-cache locality.
        if (have_next) {
            const float* prow = emb_o + (size_t)w2 * DIM;
            if (lane < 10)
                __builtin_prefetch(prow + lane * 32, 0, 3); // global_prefetch_b8
        }

        // Diagonal of 16x16 f32 C/D: lanes 0-7 -> acc[lane]; lanes 24-31 -> acc[lane-24]
        const bool b0 = (lane & 1) != 0;
        const bool b1 = (lane & 2) != 0;
        const bool b2 = (lane & 4) != 0;
        float t0 = b0 ? acc[1] : acc[0];
        float t1 = b0 ? acc[3] : acc[2];
        float t2 = b0 ? acc[5] : acc[4];
        float t3 = b0 ? acc[7] : acc[6];
        float u0 = b1 ? t1 : t0;
        float u1 = b1 ? t3 : t2;
        float sc = b2 ? u1 : u0;

        const bool valid = (lane < 8) || (lane >= 24);
        float ls = log_sigmoid_f(sgn * sc);
        if (valid) wsum += ls;
    }

    // ---- wave32 reduction ----
    #pragma unroll
    for (int off = 16; off > 0; off >>= 1)
        wsum += __shfl_xor(wsum, off, 32);
    if (lane == 0)
        partials[(blockIdx.y * gridDim.x + blockIdx.x) * WAVES + wave] = wsum;
}

__global__ __launch_bounds__(256)
void sgns_reduce_kernel(const float* __restrict__ partials, float* __restrict__ out)
{
    __shared__ float sm[256];
    float s = 0.0f;
    for (int i = threadIdx.x; i < NPART; i += 256) s += partials[i];
    sm[threadIdx.x] = s;
    __syncthreads();
    #pragma unroll
    for (int st = 128; st > 0; st >>= 1) {
        if ((int)threadIdx.x < st) sm[threadIdx.x] += sm[threadIdx.x + st];
        __syncthreads();
    }
    if (threadIdx.x == 0)
        out[0] = -sm[0] * (1.0f / (float)(BATCH * CTX));
}

extern "C" void kernel_launch(void* const* d_in, const int* in_sizes, int n_in,
                              void* d_out, int out_size, void* d_ws, size_t ws_size,
                              hipStream_t stream)
{
    const int*   iword  = (const int*)d_in[0];
    const int*   owords = (const int*)d_in[1];
    const int*   nwords = (const int*)d_in[2];
    const float* emb_i  = (const float*)d_in[3];
    const float* emb_o  = (const float*)d_in[4];
    float* out      = (float*)d_out;
    float* partials = (float*)d_ws;     // NPART floats = 32 KB scratch

    dim3 grid(BATCH / 16, YSPLIT, 1);
    sgns_score_kernel<<<grid, 256, 0, stream>>>(
        iword, owords, nwords, emb_i, emb_o, partials);
    sgns_reduce_kernel<<<1, 256, 0, stream>>>(partials, out);
}
